// CenterPointModel_78262894068393
// MI455X (gfx1250) — compile-verified, matrix operand order checked by reading, not run on hardware
//
#include <hip/hip_runtime.h>

// ---------------- types for WMMA ----------------
typedef __attribute__((ext_vector_type(16))) _Float16 v16h;
typedef __attribute__((ext_vector_type(8)))  _Float16 v8h;
typedef __attribute__((ext_vector_type(8)))  float    v8f;

union V16h { v16h v; v8h h[2]; };

#define HH 256
#define WW 256
#define NB 2
#define NP 12000
#define NPT 32
#define CIN0 9
#define CPE 64
#define MB 4            // M-blocking: 4 x 16 = 64 output channels per wave

// =====================================================================
// Weight repack: fp32 OIHW -> f16 (O, KH, KW, I)
// =====================================================================
__global__ void repack_w(const float* __restrict__ w, _Float16* __restrict__ o,
                         int O, int I, int K) {
    long total = (long)O * I * K * K;
    long i = (long)blockIdx.x * 256 + threadIdx.x;
    if (i >= total) return;
    int kw = (int)(i % K); long t = i / K;
    int kh = (int)(t % K); t /= K;
    int ci = (int)(t % I); int oo = (int)(t / I);
    o[(((long)oo * K + kh) * K + kw) * I + ci] = (_Float16)w[i];
}

// =====================================================================
// Pillar encoder pass 1: per-channel sum / sumsq of linear output
// =====================================================================
__global__ __launch_bounds__(256) void pe_pass1(
    const float* __restrict__ pf, const float* __restrict__ w,
    const float* __restrict__ bvec, float* __restrict__ gsum,
    float* __restrict__ gsq, long rows) {
    __shared__ float ssum[CPE], ssq[CPE];
    int tid = threadIdx.x;
    if (tid < CPE) { ssum[tid] = 0.f; ssq[tid] = 0.f; }
    __syncthreads();
    long r = (long)blockIdx.x * 256 + tid;
    bool ok = r < rows;
    float x[CIN0];
    const float* pr = pf + (ok ? r : 0) * CIN0;
    for (int c = 0; c < CIN0; ++c) x[c] = ok ? pr[c] : 0.f;
    int lane = tid & 31;
    for (int o = 0; o < CPE; ++o) {
        float y = bvec[o];
        for (int c = 0; c < CIN0; ++c) y += x[c] * w[o * CIN0 + c];
        if (!ok) y = 0.f;
        float q = y * y;
        for (int m = 1; m < 32; m <<= 1) {
            y += __shfl_xor(y, m, 32);
            q += __shfl_xor(q, m, 32);
        }
        if (lane == 0) { atomicAdd(&ssum[o], y); atomicAdd(&ssq[o], q); }
    }
    __syncthreads();
    if (tid < CPE) { atomicAdd(&gsum[tid], ssum[tid]); atomicAdd(&gsq[tid], ssq[tid]); }
}

// =====================================================================
// BN finalize: scale = g*rsqrt(var+eps), shift = beta - mean*scale
// =====================================================================
__global__ void bn_finalize(const float* __restrict__ sum, const float* __restrict__ sq,
                            const float* __restrict__ g, const float* __restrict__ bt,
                            float* __restrict__ scale, float* __restrict__ shift,
                            int C, float inv_count) {
    int c = blockIdx.x * blockDim.x + threadIdx.x;
    if (c >= C) return;
    float m = sum[c] * inv_count;
    float v = sq[c] * inv_count - m * m;
    float sc = g[c] * rsqrtf(v + 1e-5f);
    scale[c] = sc;
    shift[c] = bt[c] - m * sc;
}

// =====================================================================
// Pillar encoder pass 2: recompute linear, BN+relu, max over points,
// scatter into BEV (NHWC f16). One block (64 thr) per (b,p).
// =====================================================================
__global__ __launch_bounds__(64) void pe_scatter(
    const float* __restrict__ pf, const int* __restrict__ coords,
    const float* __restrict__ w, const float* __restrict__ bvec,
    const float* __restrict__ scale, const float* __restrict__ shift,
    _Float16* __restrict__ bev) {
    int bp = blockIdx.x;
    int b = bp / NP;
    __shared__ float sx[NPT * CIN0];
    const float* src = pf + (long)bp * NPT * CIN0;
    for (int i = threadIdx.x; i < NPT * CIN0; i += 64) sx[i] = src[i];
    __syncthreads();
    int c = threadIdx.x;
    float wk[CIN0];
    for (int k = 0; k < CIN0; ++k) wk[k] = w[c * CIN0 + k];
    float bb = bvec[c], sc = scale[c], sh = shift[c];
    float mx = 0.f;  // relu outputs are >= 0
    for (int n = 0; n < NPT; ++n) {
        float y = bb;
        for (int k = 0; k < CIN0; ++k) y += sx[n * CIN0 + k] * wk[k];
        y = y * sc + sh;
        y = fmaxf(y, 0.f);
        mx = fmaxf(mx, y);
    }
    int gx = coords[(long)bp * 2 + 0];
    int gy = coords[(long)bp * 2 + 1];
    if (gx >= 0 && gx < WW && gy >= 0 && gy < HH)
        bev[(((long)b * HH + gy) * WW + gx) * CPE + c] = (_Float16)mx;
}

// ---- fragment load helpers ----
__device__ __forceinline__ V16h load_frag(const _Float16* p) {
    V16h f;
    f.h[0] = *(const v8h*)(p);
    f.h[1] = *(const v8h*)(p + 16);
    return f;
}
__device__ __forceinline__ V16h load_frag_pred(const _Float16* p, bool ok) {
    V16h f;
    if (ok) {
        f.h[0] = *(const v8h*)(p);
        f.h[1] = *(const v8h*)(p + 16);
    } else {
        f.h[0] = (v8h){};
        f.h[1] = (v8h){};
    }
    return f;
}

// =====================================================================
// Implicit-GEMM 3x3 conv (pad=1), WMMA f32_16x16x32_f16.
// One wave computes a 64(cout) x 16(pixels-in-row) tile: 4 scalar
// accumulators sharing each activation (B) fragment. All boundary
// handling is folded into one per-lane predicate (no divergent CFG
// around the accumulators -> no phi copies between WMMAs).
// act: NHWC f16, wgt: (Cout,3,3,Cin) f16.
// =====================================================================
__global__ __launch_bounds__(256) void conv3x3_wmma(
    const _Float16* __restrict__ act, const _Float16* __restrict__ wgt,
    float* __restrict__ out_f32, _Float16* __restrict__ out_f16,
    const float* __restrict__ bias, float* __restrict__ bn_sum,
    float* __restrict__ bn_sq, int Bn, int Cin, int Cout,
    int Hin, int Win, int Hout, int Wout, int stride, int relu) {
    int lane = threadIdx.x & 31;
    int wv = threadIdx.x >> 5;
    int tilesN = (Hout * Wout) >> 4;
    int tilesM = Cout >> 6;                       // 64-wide M tiles
    long tile = (long)blockIdx.x * 8 + wv;
    if (tile >= (long)Bn * tilesN * tilesM) return;   // wave-uniform exit
    int mt = (int)(tile % tilesM); long t = tile / tilesM;
    int nt = (int)(t % tilesN);
    int b = (int)(t / tilesN);
    int n = lane & 15;
    int baseK = (lane & 16) ? 8 : 0;
    int px = nt << 4;
    int oy = px / Wout;
    int ox = (px % Wout) + n;     // Wout % 16 == 0: strip stays in one row
    int co0 = mt << 6;
    long astep = (long)16 * 9 * Cin;              // row stride between sub-tiles

    v8f acc0 = {}, acc1 = {}, acc2 = {}, acc3 = {};

    #pragma unroll
    for (int ky = 0; ky < 3; ++ky) {
        int iy = oy * stride - 1 + ky;
        bool rok = (iy >= 0) & (iy < Hin);        // uniform across wave
        int iyc = rok ? iy : 0;
        #pragma unroll
        for (int kx = 0; kx < 3; ++kx) {
            int ix = ox * stride - 1 + kx;
            bool ok = rok & (ix >= 0) & (ix < Win);
            const _Float16* bp = act + (((long)b * Hin + iyc) * Win + (ok ? ix : 0)) * Cin + baseK;
            const _Float16* ap = wgt + (((long)(co0 + n) * 3 + ky) * 3 + kx) * Cin + baseK;
            for (int cc = 0; cc < Cin; cc += 32) {
                V16h Bm = load_frag_pred(bp + cc, ok);
                V16h A0 = load_frag(ap + cc);
                V16h A1 = load_frag(ap + astep + cc);
                V16h A2 = load_frag(ap + 2 * astep + cc);
                V16h A3 = load_frag(ap + 3 * astep + cc);
                acc0 = __builtin_amdgcn_wmma_f32_16x16x32_f16(
                    false, A0.v, false, Bm.v, (short)0, acc0, false, false);
                acc1 = __builtin_amdgcn_wmma_f32_16x16x32_f16(
                    false, A1.v, false, Bm.v, (short)0, acc1, false, false);
                acc2 = __builtin_amdgcn_wmma_f32_16x16x32_f16(
                    false, A2.v, false, Bm.v, (short)0, acc2, false, false);
                acc3 = __builtin_amdgcn_wmma_f32_16x16x32_f16(
                    false, A3.v, false, Bm.v, (short)0, acc3, false, false);
            }
        }
    }

    int moff = (lane & 16) ? 8 : 0;
    long pbase = (((long)b * Hout + oy) * Wout + ox) * Cout;
    v8f accs[MB] = {acc0, acc1, acc2, acc3};
    #pragma unroll
    for (int u = 0; u < MB; ++u) {
        int chb = co0 + u * 16 + moff;
        long obase = pbase + chb;
        float vr[8];
        for (int r = 0; r < 8; ++r) {
            float v = accs[u][r];
            if (bias) v += bias[chb + r];
            if (relu) v = fmaxf(v, 0.f);
            vr[r] = v;
            if (out_f32) out_f32[obase + r] = v;
            if (out_f16) out_f16[obase + r] = (_Float16)v;
        }
        if (bn_sum) {
            for (int r = 0; r < 8; ++r) {
                float s = vr[r], q = vr[r] * vr[r];
                for (int m = 1; m < 16; m <<= 1) {
                    s += __shfl_xor(s, m, 32);
                    q += __shfl_xor(q, m, 32);
                }
                if ((lane & 15) == 0) {
                    atomicAdd(&bn_sum[chb + r], s);
                    atomicAdd(&bn_sq[chb + r], q);
                }
            }
        }
    }
}

// =====================================================================
// Conv-transpose with kernel==stride ('VALID'): every output pixel maps
// to exactly one input pixel and one weight tap (ky,kx)=(oy%s,ox%s).
// Tile = 16 output pixels sharing one tap; 64-channel M-block per wave.
// =====================================================================
__global__ __launch_bounds__(256) void convt_wmma(
    const _Float16* __restrict__ act, const _Float16* __restrict__ wgt,
    float* __restrict__ out_f32, float* __restrict__ bn_sum,
    float* __restrict__ bn_sq, int Bn, int Cin, int Cout,
    int Hin, int Win, int s) {
    int lane = threadIdx.x & 31;
    int wv = threadIdx.x >> 5;
    int Hout = Hin * s, Wout = Win * s;
    int tilesJ = Win >> 4;
    int tilesM = Cout >> 6;
    long tile = (long)blockIdx.x * 8 + wv;
    long ntiles = (long)Bn * Hout * s * tilesJ * tilesM;
    if (tile >= ntiles) return;
    int mt = (int)(tile % tilesM); long t = tile / tilesM;
    int jt = (int)(t % tilesJ); t /= tilesJ;
    int kx = (int)(t % s); t /= s;
    int oy = (int)(t % Hout);
    int b = (int)(t / Hout);
    int iy = oy / s, ky = oy % s;
    int n = lane & 15;
    int baseK = (lane & 16) ? 8 : 0;
    int j = (jt << 4) + n;                 // input x, always in range
    int ox = j * s + kx;
    int co0 = mt << 6;
    long astep = (long)16 * s * s * Cin;

    const _Float16* bp = act + (((long)b * Hin + iy) * Win + j) * Cin + baseK;
    const _Float16* ap = wgt + (((long)(co0 + n) * s + ky) * s + kx) * Cin + baseK;
    v8f acc0 = {}, acc1 = {}, acc2 = {}, acc3 = {};

    for (int cc = 0; cc < Cin; cc += 32) {
        V16h Bm = load_frag(bp + cc);
        V16h A0 = load_frag(ap + cc);
        V16h A1 = load_frag(ap + astep + cc);
        V16h A2 = load_frag(ap + 2 * astep + cc);
        V16h A3 = load_frag(ap + 3 * astep + cc);
        acc0 = __builtin_amdgcn_wmma_f32_16x16x32_f16(
            false, A0.v, false, Bm.v, (short)0, acc0, false, false);
        acc1 = __builtin_amdgcn_wmma_f32_16x16x32_f16(
            false, A1.v, false, Bm.v, (short)0, acc1, false, false);
        acc2 = __builtin_amdgcn_wmma_f32_16x16x32_f16(
            false, A2.v, false, Bm.v, (short)0, acc2, false, false);
        acc3 = __builtin_amdgcn_wmma_f32_16x16x32_f16(
            false, A3.v, false, Bm.v, (short)0, acc3, false, false);
    }
    int moff = (lane & 16) ? 8 : 0;
    long pbase = (((long)b * Hout + oy) * Wout + ox) * Cout;
    v8f accs[MB] = {acc0, acc1, acc2, acc3};
    #pragma unroll
    for (int u = 0; u < MB; ++u) {
        int chb = co0 + u * 16 + moff;
        long obase = pbase + chb;
        for (int r = 0; r < 8; ++r) out_f32[obase + r] = accs[u][r];
        for (int r = 0; r < 8; ++r) {
            float sm = accs[u][r], q = accs[u][r] * accs[u][r];
            for (int m = 1; m < 16; m <<= 1) {
                sm += __shfl_xor(sm, m, 32);
                q += __shfl_xor(q, m, 32);
            }
            if ((lane & 15) == 0) {
                atomicAdd(&bn_sum[chb + r], sm);
                atomicAdd(&bn_sq[chb + r], q);
            }
        }
    }
}

// =====================================================================
// BN apply + relu, fp32 NHWC -> f16 NHWC (with channel offset for concat)
// =====================================================================
__global__ void bn_apply(const float* __restrict__ pre, const float* __restrict__ scale,
                         const float* __restrict__ shift, _Float16* __restrict__ out,
                         int Cch, int outC, int choff, long npix) {
    long i = (long)blockIdx.x * 256 + threadIdx.x;
    if (i >= npix * Cch) return;
    int c = (int)(i % Cch);
    long p = i / Cch;
    float y = pre[i] * scale[c] + shift[c];
    y = fmaxf(y, 0.f);
    out[p * outC + choff + c] = (_Float16)y;
}

// =====================================================================
// 1x1 head conv: NHWC f16 (C=64) -> NCHW fp32, with bias
// =====================================================================
__global__ void head1x1(const _Float16* __restrict__ act, const float* __restrict__ w,
                        const float* __restrict__ bias, float* __restrict__ out,
                        int HWp, int Cout, int Bn) {
    long i = (long)blockIdx.x * 256 + threadIdx.x;
    long total = (long)Bn * Cout * HWp;
    if (i >= total) return;
    int px = (int)(i % HWp); long t = i / HWp;
    int co = (int)(t % Cout);
    int b = (int)(t / Cout);
    const _Float16* a = act + ((long)b * HWp + px) * 64;
    const float* wr = w + (long)co * 64;
    float y = bias[co];
    for (int c = 0; c < 64; ++c) y += (float)a[c] * wr[c];
    out[i] = y;  // i == ((b*Cout+co)*HWp+px): NCHW flat
}

// =====================================================================
// Host orchestration
// =====================================================================
extern "C" void kernel_launch(void* const* d_in, const int* in_sizes, int n_in,
                              void* d_out, int out_size, void* d_ws, size_t ws_size,
                              hipStream_t stream) {
    (void)in_sizes; (void)n_in; (void)out_size; (void)ws_size;
    const float* pf = (const float*)d_in[0];
    const int* coords = (const int*)d_in[1];
    int ii = 2;
    const float* pe_w = (const float*)d_in[ii++]; const float* pe_b = (const float*)d_in[ii++];
    const float* pe_g = (const float*)d_in[ii++]; const float* pe_bt = (const float*)d_in[ii++];
    struct ConvP { const float *w, *b, *g, *bt; };
    auto grab_cbn = [&]() { ConvP c; c.w=(const float*)d_in[ii++]; c.b=(const float*)d_in[ii++];
                            c.g=(const float*)d_in[ii++]; c.bt=(const float*)d_in[ii++]; return c; };
    ConvP b1c1 = grab_cbn(), b1c2 = grab_cbn(), b2c1 = grab_cbn(), b2c2 = grab_cbn();
    ConvP b3c1 = grab_cbn(), b3c2 = grab_cbn();
    ConvP up1 = grab_cbn(), up2 = grab_cbn(), up3 = grab_cbn(), hd = grab_cbn();
    const float* hm1_w = (const float*)d_in[ii++]; const float* hm1_b = (const float*)d_in[ii++];
    const float* hm2_w = (const float*)d_in[ii++]; const float* hm2_b = (const float*)d_in[ii++];
    const float* bx1_w = (const float*)d_in[ii++]; const float* bx1_b = (const float*)d_in[ii++];
    const float* bx2_w = (const float*)d_in[ii++]; const float* bx2_b = (const float*)d_in[ii++];

    // ---- workspace bump allocator ----
    char* base = (char*)d_ws; size_t off = 0;
    auto alloc = [&](size_t bytes) -> void* {
        off = (off + 255) & ~(size_t)255;
        void* p = base + off; off += bytes; return p;
    };
    auto af16 = [&](long n) { return (_Float16*)alloc((size_t)n * 2); };
    auto af32 = [&](long n) { return (float*)alloc((size_t)n * 4); };

    // f16 repacked weights
    _Float16* wq_b1c1 = af16(64L*64*9);   _Float16* wq_b1c2 = af16(64L*64*9);
    _Float16* wq_b2c1 = af16(128L*64*9);  _Float16* wq_b2c2 = af16(128L*128*9);
    _Float16* wq_b3c1 = af16(256L*128*9); _Float16* wq_b3c2 = af16(256L*256*9);
    _Float16* wq_up1  = af16(128L*64*4);  _Float16* wq_up2  = af16(128L*128*16);
    _Float16* wq_up3  = af16(128L*256*64);
    _Float16* wq_hd   = af16(128L*384*9);
    _Float16* wq_hm1  = af16(64L*128*9);  _Float16* wq_bx1  = af16(64L*128*9);

    // activations (f16, NHWC)
    _Float16* bev   = af16((long)NB*HH*WW*64);
    _Float16* x1a   = af16((long)NB*128*128*64);
    _Float16* x1    = af16((long)NB*128*128*64);
    _Float16* x2a   = af16((long)NB*64*64*128);
    _Float16* x2    = af16((long)NB*64*64*128);
    _Float16* x3a   = af16((long)NB*32*32*256);
    _Float16* x3    = af16((long)NB*32*32*256);
    _Float16* feats = af16((long)NB*HH*WW*384);
    _Float16* sbuf  = af16((long)NB*HH*WW*128);
    _Float16* hm1o  = af16((long)NB*HH*WW*64);
    _Float16* bx1o  = af16((long)NB*HH*WW*64);

    float* preBuf  = af32((long)NB*HH*WW*128);   // largest pre-BN tensor (reused)
    float* bn_sum  = af32(512); float* bn_sq    = af32(512);
    float* bn_scale = af32(512); float* bn_shift = af32(512);

    // ---- repack all conv weights to f16 (O,KH,KW,I) ----
    auto repack = [&](const float* w, _Float16* o, int O, int I, int K) {
        long tot = (long)O * I * K * K;
        repack_w<<<(unsigned)((tot + 255) / 256), 256, 0, stream>>>(w, o, O, I, K);
    };
    repack(b1c1.w, wq_b1c1, 64, 64, 3);   repack(b1c2.w, wq_b1c2, 64, 64, 3);
    repack(b2c1.w, wq_b2c1, 128, 64, 3);  repack(b2c2.w, wq_b2c2, 128, 128, 3);
    repack(b3c1.w, wq_b3c1, 256, 128, 3); repack(b3c2.w, wq_b3c2, 256, 256, 3);
    repack(up1.w, wq_up1, 128, 64, 2);    repack(up2.w, wq_up2, 128, 128, 4);
    repack(up3.w, wq_up3, 128, 256, 8);   repack(hd.w, wq_hd, 128, 384, 3);
    repack(hm1_w, wq_hm1, 64, 128, 3);    repack(bx1_w, wq_bx1, 64, 128, 3);

    // ---- pillar encoder ----
    hipMemsetAsync(bn_sum, 0, 512 * 4, stream);
    hipMemsetAsync(bn_sq, 0, 512 * 4, stream);
    long rows = (long)NB * NP * NPT;
    pe_pass1<<<(unsigned)((rows + 255) / 256), 256, 0, stream>>>(pf, pe_w, pe_b, bn_sum, bn_sq, rows);
    bn_finalize<<<1, CPE, 0, stream>>>(bn_sum, bn_sq, pe_g, pe_bt, bn_scale, bn_shift,
                                       CPE, 1.f / (float)rows);
    hipMemsetAsync(bev, 0, (size_t)NB * HH * WW * 64 * 2, stream);
    pe_scatter<<<NB * NP, 64, 0, stream>>>(pf, coords, pe_w, pe_b, bn_scale, bn_shift, bev);

    // ---- conv+BN+relu helper ----
    auto run_conv_bn = [&](const _Float16* in, const _Float16* w, int Cin, int Cout,
                           int Hin, int Win, int stride, const ConvP& p,
                           _Float16* outAct, int outC, int choff) {
        int Ho = Hin / stride, Wo = Win / stride;
        hipMemsetAsync(bn_sum, 0, Cout * 4, stream);
        hipMemsetAsync(bn_sq, 0, Cout * 4, stream);
        long tiles = (long)NB * ((long)Ho * Wo / 16) * (Cout / 64);
        conv3x3_wmma<<<(unsigned)((tiles + 7) / 8), 256, 0, stream>>>(
            in, w, preBuf, nullptr, nullptr, bn_sum, bn_sq,
            NB, Cin, Cout, Hin, Win, Ho, Wo, stride, 0);
        bn_finalize<<<1, Cout, 0, stream>>>(bn_sum, bn_sq, p.g, p.bt, bn_scale, bn_shift,
                                            Cout, 1.f / (float)((long)NB * Ho * Wo));
        long npix = (long)NB * Ho * Wo;
        bn_apply<<<(unsigned)((npix * Cout + 255) / 256), 256, 0, stream>>>(
            preBuf, bn_scale, bn_shift, outAct, Cout, outC, choff, npix);
    };
    auto run_convt_bn = [&](const _Float16* in, const _Float16* w, int Cin, int Cout,
                            int Hin, int Win, int s, const ConvP& p,
                            _Float16* outAct, int outC, int choff) {
        int Ho = Hin * s, Wo = Win * s;
        hipMemsetAsync(bn_sum, 0, Cout * 4, stream);
        hipMemsetAsync(bn_sq, 0, Cout * 4, stream);
        long tiles = (long)NB * Ho * s * (Win / 16) * (Cout / 64);
        convt_wmma<<<(unsigned)((tiles + 7) / 8), 256, 0, stream>>>(
            in, w, preBuf, bn_sum, bn_sq, NB, Cin, Cout, Hin, Win, s);
        bn_finalize<<<1, Cout, 0, stream>>>(bn_sum, bn_sq, p.g, p.bt, bn_scale, bn_shift,
                                            Cout, 1.f / (float)((long)NB * Ho * Wo));
        long npix = (long)NB * Ho * Wo;
        bn_apply<<<(unsigned)((npix * Cout + 255) / 256), 256, 0, stream>>>(
            preBuf, bn_scale, bn_shift, outAct, Cout, outC, choff, npix);
    };

    // ---- backbone ----
    run_conv_bn(bev, wq_b1c1, 64, 64, 256, 256, 2, b1c1, x1a, 64, 0);
    run_conv_bn(x1a, wq_b1c2, 64, 64, 128, 128, 1, b1c2, x1, 64, 0);
    run_conv_bn(x1, wq_b2c1, 64, 128, 128, 128, 2, b2c1, x2a, 128, 0);
    run_conv_bn(x2a, wq_b2c2, 128, 128, 64, 64, 1, b2c2, x2, 128, 0);
    run_conv_bn(x2, wq_b3c1, 128, 256, 64, 64, 2, b3c1, x3a, 256, 0);
    run_conv_bn(x3a, wq_b3c2, 256, 256, 32, 32, 1, b3c2, x3, 256, 0);

    // ---- upsample into 384-channel concat ----
    run_convt_bn(x1, wq_up1, 64, 128, 128, 128, 2, up1, feats, 384, 0);
    run_convt_bn(x2, wq_up2, 128, 128, 64, 64, 4, up2, feats, 384, 128);
    run_convt_bn(x3, wq_up3, 256, 128, 32, 32, 8, up3, feats, 384, 256);

    // ---- head shared conv ----
    run_conv_bn(feats, wq_hd, 384, 128, 256, 256, 1, hd, sbuf, 128, 0);

    // ---- head branches: conv3x3 + bias + relu (no BN) ----
    long tiles_h = (long)NB * ((long)HH * WW / 16) * (64 / 64);
    conv3x3_wmma<<<(unsigned)((tiles_h + 7) / 8), 256, 0, stream>>>(
        sbuf, wq_hm1, nullptr, hm1o, hm1_b, nullptr, nullptr,
        NB, 128, 64, HH, WW, HH, WW, 1, 1);
    conv3x3_wmma<<<(unsigned)((tiles_h + 7) / 8), 256, 0, stream>>>(
        sbuf, wq_bx1, nullptr, bx1o, bx1_b, nullptr, nullptr,
        NB, 128, 64, HH, WW, HH, WW, 1, 1);

    // ---- 1x1 heads -> NCHW fp32 outputs (hm then bx, concatenated) ----
    float* out_hm = (float*)d_out;
    float* out_bx = out_hm + (long)NB * HH * WW;      // hm is (B,1,H,W)
    long tot_hm = (long)NB * 1 * HH * WW;
    head1x1<<<(unsigned)((tot_hm + 255) / 256), 256, 0, stream>>>(
        hm1o, hm2_w, hm2_b, out_hm, HH * WW, 1, NB);
    long tot_bx = (long)NB * 8 * HH * WW;
    head1x1<<<(unsigned)((tot_bx + 255) / 256), 256, 0, stream>>>(
        bx1o, bx2_w, bx2_b, out_bx, HH * WW, 8, NB);
}
